// WPHModel_33036888441016
// MI455X (gfx1250) — matrix operand order repaired and checked by reading, not compiled, not run on hardware
//
#include <hip/hip_runtime.h>
#include <math.h>

#define MDIM   128
#define NPIX   16384
#define NBATCH 2
#define NF     80        // C*J*L*A = 1*5*4*4
#define NSHIFT 5
#define NPAIR  960
#define OUT_PER_B 4806   // 4800 corr + 2 low + 4 high
#define PI_F 3.14159265358979323846f

typedef __bf16 bf16_t;
typedef __attribute__((ext_vector_type(16))) __bf16 v16bf;
typedef __attribute__((ext_vector_type(8)))  __bf16 v8bf;
typedef __attribute__((ext_vector_type(8)))  float  v8f;

union frag16 {            // build a 16-wide WMMA fragment from two 16B loads
  v16bf v;
  v8bf  h[2];
};

__device__ __forceinline__ int bitrev7(int v) {
  return (int)(__brev((unsigned)v) >> 25);
}

__constant__ int c_dx[NSHIFT] = {0, 4, 3, 0, -3};
__constant__ int c_dy[NSHIFT] = {0, 0, 3, 4, 3};

// ---------------------------------------------------------------------------
// 128-point radix-2 FFT stages over a 128x128 LDS-resident complex field.
// Forward: DIF (natural in -> bit-reversed out). Inverse: DIT (bit-reversed
// in -> natural out). Pointwise products in between use bitrev addressing of
// the natural-order filter arrays, so no data permutation is ever required.
// 8192 butterflies per stage, strided over the workgroup.
// ---------------------------------------------------------------------------
__device__ __forceinline__ void fft_stage_dif(float* re, float* im, int tid,
                                              int nt, bool cols, int ls) {
  const int span = 1 << ls;
  for (int j = tid; j < 8192; j += nt) {
    int line, bf;
    if (!cols) { line = j >> 6;  bf = j & 63; }
    else       { line = j & 127; bf = j >> 7; }
    const int grp = bf >> ls;
    const int pos = bf & (span - 1);
    const int base = grp * (span << 1) + pos;
    int i0, i1;
    if (!cols) { i0 = line * MDIM + base; i1 = i0 + span; }
    else       { i0 = base * MDIM + line; i1 = i0 + span * MDIM; }
    const float ar = re[i0], ai = im[i0];
    const float br = re[i1], bi = im[i1];
    re[i0] = ar + br; im[i0] = ai + bi;
    const float tr = ar - br, ti = ai - bi;
    float sn, cs;
    __sincosf(-PI_F * (float)pos / (float)span, &sn, &cs);
    re[i1] = tr * cs - ti * sn;
    im[i1] = tr * sn + ti * cs;
  }
}

__device__ __forceinline__ void fft_stage_dit(float* re, float* im, int tid,
                                              int nt, bool cols, int ls) {
  const int span = 1 << ls;
  for (int j = tid; j < 8192; j += nt) {
    int line, bf;
    if (!cols) { line = j >> 6;  bf = j & 63; }
    else       { line = j & 127; bf = j >> 7; }
    const int grp = bf >> ls;
    const int pos = bf & (span - 1);
    const int base = grp * (span << 1) + pos;
    int i0, i1;
    if (!cols) { i0 = line * MDIM + base; i1 = i0 + span; }
    else       { i0 = base * MDIM + line; i1 = i0 + span * MDIM; }
    const float ar = re[i0], ai = im[i0];
    const float br = re[i1], bi = im[i1];
    float sn, cs;
    __sincosf(PI_F * (float)pos / (float)span, &sn, &cs);  // inverse: +angle
    const float wr = br * cs - bi * sn;
    const float wi = br * sn + bi * cs;
    re[i0] = ar + wr; im[i0] = ai + wi;
    re[i1] = ar - wr; im[i1] = ai - wi;
  }
}

__device__ void fft2d_dif(float* re, float* im, int tid, int nt) {
  for (int ls = 6; ls >= 0; --ls) { fft_stage_dif(re, im, tid, nt, false, ls); __syncthreads(); }
  for (int ls = 6; ls >= 0; --ls) { fft_stage_dif(re, im, tid, nt, true,  ls); __syncthreads(); }
}
__device__ void fft2d_dit_inv(float* re, float* im, int tid, int nt) {
  for (int ls = 0; ls <= 6; ++ls) { fft_stage_dit(re, im, tid, nt, false, ls); __syncthreads(); }
  for (int ls = 0; ls <= 6; ++ls) { fft_stage_dit(re, im, tid, nt, true,  ls); __syncthreads(); }
}

// ---------------------------------------------------------------------------
// Kernel A: hatx = fft2(x), stored bit-reversed along both axes.
// ---------------------------------------------------------------------------
__global__ void kfft_fwd(const float* __restrict__ x, float* __restrict__ hatr,
                         float* __restrict__ hati) {
  extern __shared__ float sm[];
  float* re = sm;
  float* im = sm + NPIX;
  const int b = blockIdx.x, tid = threadIdx.x, nt = blockDim.x;
  for (int p = tid; p < NPIX; p += nt) { re[p] = x[b * NPIX + p]; im[p] = 0.f; }
  __syncthreads();
  fft2d_dif(re, im, tid, nt);
  for (int p = tid; p < NPIX; p += nt) {
    hatr[b * NPIX + p] = re[p];
    hati[b * NPIX + p] = im[p];
  }
}

// ---------------------------------------------------------------------------
// Kernel B (fused): hatx*hatpsi[f] -> ifft2 -> relu -> mean/std normalize ->
// bf16, stored as 5 pre-rolled copies Ys[s][b][f][pix] = y[shift_s(pix)]
// so the Gram GEMM reads both operands with contiguous 16B vector loads.
// One workgroup per (f, b); full field resident in 128KB LDS.
// ---------------------------------------------------------------------------
__global__ void kfilter(const float* __restrict__ hpsir, const float* __restrict__ hpsii,
                        const float* __restrict__ hatr,  const float* __restrict__ hati,
                        bf16_t* __restrict__ Ys) {
  extern __shared__ float sm[];
  float* re = sm;
  float* im = sm + NPIX;
  float* r1 = sm + 2 * NPIX;
  float* r2 = r1 + 256;
  const int f = blockIdx.x, b = blockIdx.y;
  const int tid = threadIdx.x, nt = blockDim.x;

  for (int p = tid; p < NPIX; p += nt) {
    const int nr = bitrev7(p >> 7), nc = bitrev7(p & 127);
    const float hr = hpsir[f * NPIX + nr * MDIM + nc];
    const float hi = hpsii[f * NPIX + nr * MDIM + nc];
    const float xr = hatr[b * NPIX + p], xi = hati[b * NPIX + p];
    re[p] = xr * hr - xi * hi;
    im[p] = xr * hi + xi * hr;
  }
  __syncthreads();
  fft2d_dit_inv(re, im, tid, nt);

  const float scale = 1.0f / 16384.0f;   // ifft2 normalization
  float s1 = 0.f, s2 = 0.f;
  for (int p = tid; p < NPIX; p += nt) {
    float v = re[p] * scale;
    v = v > 0.f ? v : 0.f;
    s1 += v; s2 += v * v;
  }
  r1[tid] = s1; r2[tid] = s2;
  __syncthreads();
  for (int o = nt >> 1; o > 0; o >>= 1) {
    if (tid < o) { r1[tid] += r1[tid + o]; r2[tid] += r2[tid + o]; }
    __syncthreads();
  }
  const float mean = r1[0] * (1.0f / NPIX);
  float var = r2[0] * (1.0f / NPIX) - mean * mean;
  var = var > 0.f ? var : 0.f;
  const float inv = 1.0f / (sqrtf(var) + 1e-6f);

  const size_t plane = (size_t)NBATCH * NF * NPIX;     // one shift's worth
  bf16_t* yo = Ys + ((size_t)b * NF + f) * NPIX;       // base of shift 0 map
  for (int p = tid; p < NPIX; p += nt) {
    float v = re[p] * scale;
    v = v > 0.f ? v : 0.f;
    const bf16_t q = (bf16_t)((v - mean) * inv);
    const int x = p >> 7, y = p & 127;
    // Ys[s][..][pix] = y[shift_s(pix)]  with shift_s(px,py)=((px-dx),(py-dy))
    // so source pixel p lands at destination ((x+dx)&127, (y+dy)&127).
    #pragma unroll
    for (int s = 0; s < NSHIFT; ++s) {
      const int dxp = (x + c_dx[s]) & 127;
      const int dyp = (y + c_dy[s]) & 127;
      yo[(size_t)s * plane + dxp * MDIM + dyp] = q;
    }
  }
}

// ---------------------------------------------------------------------------
// Kernel C: shifted 80x80 Gram matrices G[b,s] = Y * shift_s(Y)^T via
// v_wmma_f32_16x16x32_bf16. One wave per (b, s, 16x16 tile); K = 16384,
// unrolled x2 with independent accumulators. All fragment loads are
// contiguous 16B chunks matching the ISA 16-bit A/B VGPR layouts.
// ---------------------------------------------------------------------------
__global__ void kgram(const bf16_t* __restrict__ Ys, float* __restrict__ G) {
  const int blk  = blockIdx.x;
  const int tile = blk % 25;
  const int rest = blk / 25;
  const int s = rest % NSHIFT;
  const int b = rest / NSHIFT;
  const int tm = tile / 5, tn = tile % 5;
  const int lane = threadIdx.x & 31;
  const int half = lane >> 4;          // K-half select (ISA 16-bit layouts)
  const int l15  = lane & 15;

  const size_t plane = (size_t)NBATCH * NF * NPIX;
  // A: unshifted maps (shift 0 plane), row m1 = tm*16 + l15
  const bf16_t* Ya = Ys + ((size_t)b * NF + (tm * 16 + l15)) * NPIX;
  // B: pre-rolled maps for shift s, col m2 = tn*16 + l15
  const bf16_t* Yb = Ys + (size_t)s * plane + ((size_t)b * NF + (tn * 16 + l15)) * NPIX;

  v8f acc0 = {}, acc1 = {};
  for (int k0 = 0; k0 < NPIX; k0 += 64) {
    __builtin_prefetch(Ya + k0 + 1024, 0, 1);   // global_prefetch_b8
    __builtin_prefetch(Yb + k0 + 1024, 0, 1);
    #pragma unroll
    for (int u = 0; u < 2; ++u) {
      const int kb = k0 + 32 * u;
      frag16 af, bf;
      // A 16x32 bf16 layout: lane = (row l15, K-half): VGPR0-3 hold
      // K = 8*half+[0,8), VGPR4-7 hold K = 16+8*half+[0,8).
      af.h[0] = *(const v8bf*)(Ya + kb + 8 * half);
      af.h[1] = *(const v8bf*)(Ya + kb + 16 + 8 * half);
      // B 32x16 bf16 layout: lane = (col l15, K-half): VGPR0-3 hold
      // K = 8*half+[0,8), VGPR4-7 hold K = 16+8*half+[0,8).
      bf.h[0] = *(const v8bf*)(Yb + kb + 8 * half);
      bf.h[1] = *(const v8bf*)(Yb + kb + 16 + 8 * half);
      if (u == 0)
        acc0 = __builtin_amdgcn_wmma_f32_16x16x32_bf16(
            false, af.v, false, bf.v, (short)0, acc0, false, false);
      else
        acc1 = __builtin_amdgcn_wmma_f32_16x16x32_bf16(
            false, af.v, false, bf.v, (short)0, acc1, false, false);
    }
  }

  // C/D layout: VGPR r -> M = r + 8*half, N = l15.  80 = 5*16 exactly.
  const float scale = 1.0f / 16384.0f;  // einsum / (M*N)
  float* Gt = G + ((size_t)b * NSHIFT + s) * (NF * NF);
  #pragma unroll
  for (int r = 0; r < 8; ++r) {
    const int row = tm * 16 + r + half * 8;
    const int col = tn * 16 + l15;
    Gt[row * NF + col] = (acc0[r] + acc1[r]) * scale;
  }
}

// ---------------------------------------------------------------------------
// Kernel D: gather xcorr entries; pair indices decoded arithmetically
// (c=0; j1<=j2<=4; all (l1,l2) pass delta_l=4; a1 in [0,4), a2=0).
// ---------------------------------------------------------------------------
__global__ void kgather(const float* __restrict__ G, float* __restrict__ out) {
  const int t = blockIdx.x * blockDim.x + threadIdx.x;
  if (t >= NBATCH * NSHIFT * NPAIR) return;
  const int p = t % NPAIR;
  const int rest = t / NPAIR;
  const int s = rest % NSHIFT;
  const int b = rest / NSHIFT;
  int jj = p >> 6;
  const int rem = p & 63;
  const int l1 = rem >> 4, l2 = (rem >> 2) & 3, a1 = rem & 3;
  int j1 = 0, cnt = 5;
  while (jj >= cnt) { jj -= cnt; ++j1; --cnt; }
  const int j2 = j1 + jj;
  const int i1 = (j1 * 4 + l1) * 4 + a1;
  const int i2 = (j2 * 4 + l2) * 4;
  out[b * OUT_PER_B + s * NPAIR + p] =
      G[(((size_t)b * NSHIFT + s) * NF + i1) * NF + i2];
}

// ---------------------------------------------------------------------------
// Kernel E: xlow = [mean, mean(sq)] of ifft2(hatx*hatphi).real
// ---------------------------------------------------------------------------
__global__ void kxlow(const float* __restrict__ hphir, const float* __restrict__ hphii,
                      const float* __restrict__ hatr,  const float* __restrict__ hati,
                      float* __restrict__ out) {
  extern __shared__ float sm[];
  float* re = sm;
  float* im = sm + NPIX;
  float* r1 = sm + 2 * NPIX;
  float* r2 = r1 + 256;
  const int b = blockIdx.x, tid = threadIdx.x, nt = blockDim.x;

  for (int p = tid; p < NPIX; p += nt) {
    const int nr = bitrev7(p >> 7), nc = bitrev7(p & 127);
    const float hr = hphir[nr * MDIM + nc];
    const float hi = hphii[nr * MDIM + nc];
    const float xr = hatr[b * NPIX + p], xi = hati[b * NPIX + p];
    re[p] = xr * hr - xi * hi;
    im[p] = xr * hi + xi * hr;
  }
  __syncthreads();
  fft2d_dit_inv(re, im, tid, nt);

  const float scale = 1.0f / 16384.0f;
  float s1 = 0.f, s2 = 0.f;
  for (int p = tid; p < NPIX; p += nt) {
    const float v = re[p] * scale;
    s1 += v; s2 += v * v;
  }
  r1[tid] = s1; r2[tid] = s2;
  __syncthreads();
  for (int o = nt >> 1; o > 0; o >>= 1) {
    if (tid < o) { r1[tid] += r1[tid + o]; r2[tid] += r2[tid + o]; }
    __syncthreads();
  }
  if (tid == 0) {
    out[b * OUT_PER_B + 4800] = r1[0] * (1.0f / NPIX);
    out[b * OUT_PER_B + 4801] = r2[0] * (1.0f / NPIX);
  }
}

// ---------------------------------------------------------------------------
// Kernel F: xhigh — Fourier-domain power in 4 angular highpass sectors,
// masks computed on the fly (|k| > pi/2, theta folded to [0,pi)).
// ---------------------------------------------------------------------------
__global__ void kxhigh(const float* __restrict__ hatr, const float* __restrict__ hati,
                       float* __restrict__ out) {
  __shared__ float red[4][256];
  const int b = blockIdx.x, tid = threadIdx.x, nt = blockDim.x;
  float acc[4] = {0.f, 0.f, 0.f, 0.f};
  for (int p = tid; p < NPIX; p += nt) {
    const int nr = bitrev7(p >> 7), nc = bitrev7(p & 127);  // natural freq idx
    const float fx = (float)(nr < 64 ? nr : nr - 128) * (1.0f / 128.0f);
    const float fy = (float)(nc < 64 ? nc : nc - 128) * (1.0f / 128.0f);
    const float kx = 2.0f * PI_F * fx, ky = 2.0f * PI_F * fy;
    const float rm = sqrtf(kx * kx + ky * ky);
    if (rm > 0.5f * PI_F) {
      float th = atan2f(ky, kx);
      th = th - floorf(th / PI_F) * PI_F;        // mod pi -> [0, pi)
      int s = (int)(th / (PI_F * 0.25f));
      s = s > 3 ? 3 : (s < 0 ? 0 : s);
      const float xr = hatr[b * NPIX + p], xi = hati[b * NPIX + p];
      acc[s] += xr * xr + xi * xi;
    }
  }
  #pragma unroll
  for (int s = 0; s < 4; ++s) red[s][tid] = acc[s];
  __syncthreads();
  for (int o = nt >> 1; o > 0; o >>= 1) {
    if (tid < o) {
      #pragma unroll
      for (int s = 0; s < 4; ++s) red[s][tid] += red[s][tid + o];
    }
    __syncthreads();
  }
  if (tid < 4)
    out[b * OUT_PER_B + 4802 + tid] = red[tid][0] * (1.0f / (16384.0f * 16384.0f));
}

// ---------------------------------------------------------------------------
extern "C" void kernel_launch(void* const* d_in, const int* in_sizes, int n_in,
                              void* d_out, int out_size, void* d_ws, size_t ws_size,
                              hipStream_t stream) {
  const float* x     = (const float*)d_in[0];
  const float* hpsir = (const float*)d_in[1];
  const float* hpsii = (const float*)d_in[2];
  const float* hphir = (const float*)d_in[3];
  const float* hphii = (const float*)d_in[4];
  float* out = (float*)d_out;

  // Workspace layout (~26.7 MB): hatx re/im, 5 pre-rolled bf16 map sets, Gram.
  float*  hatr = (float*)d_ws;                               // NBATCH*NPIX
  float*  hati = hatr + (size_t)NBATCH * NPIX;               // NBATCH*NPIX
  bf16_t* Ys   = (bf16_t*)(hati + (size_t)NBATCH * NPIX);    // NSHIFT*NBATCH*NF*NPIX
  float*  G    = (float*)((char*)Ys +
                 (size_t)NSHIFT * NBATCH * NF * NPIX * sizeof(bf16_t));

  const size_t smFFT  = (size_t)2 * NPIX * sizeof(float);          // 128 KB
  const size_t smFilt = ((size_t)2 * NPIX + 512) * sizeof(float);  // 130 KB

  // Allow >64KB dynamic LDS (CDNA5 WGP: up to 320KB/workgroup). Ignore errors.
  (void)hipFuncSetAttribute((const void*)kfft_fwd,
      hipFuncAttributeMaxDynamicSharedMemorySize, (int)smFFT);
  (void)hipFuncSetAttribute((const void*)kfilter,
      hipFuncAttributeMaxDynamicSharedMemorySize, (int)smFilt);
  (void)hipFuncSetAttribute((const void*)kxlow,
      hipFuncAttributeMaxDynamicSharedMemorySize, (int)smFilt);

  kfft_fwd<<<NBATCH, 256, smFFT, stream>>>(x, hatr, hati);
  kfilter<<<dim3(NF, NBATCH), 256, smFilt, stream>>>(hpsir, hpsii, hatr, hati, Ys);
  kgram<<<NBATCH * NSHIFT * 25, 32, 0, stream>>>(Ys, G);
  kgather<<<(NBATCH * NSHIFT * NPAIR + 255) / 256, 256, 0, stream>>>(G, out);
  kxlow<<<NBATCH, 256, smFilt, stream>>>(hphir, hphii, hatr, hati, out);
  kxhigh<<<NBATCH, 256, 0, stream>>>(hatr, hati, out);
}
